// RelationAwareAspectExtractor_15006615733558
// MI455X (gfx1250) — matrix operand order repaired and checked by reading, not compiled, not run on hardware
//
#include <hip/hip_runtime.h>
#include <hip/hip_bf16.h>

typedef __bf16 bf16_t;
typedef __bf16 v16bf __attribute__((ext_vector_type(16)));
typedef __bf16 v8bf  __attribute__((ext_vector_type(8)));
typedef float  v8f   __attribute__((ext_vector_type(8)));

#define B_ 16
#define S_ 1024
#define E_ 256
#define H_ 256
#define N4H 1024   // 4*H
#define MROWS (S_*B_)   // 16384

// ---------------------------------------------------------------------------
// Embedding gather: x_bf16[(s*B+b)*E + e] = emb[ids[b*S+s]*E + e]
// ---------------------------------------------------------------------------
__global__ void k_embed(const int* __restrict__ ids,
                        const float* __restrict__ emb,
                        bf16_t* __restrict__ x) {
    int idx = blockIdx.x * blockDim.x + threadIdx.x;
    if (idx >= S_ * B_ * E_) return;
    int e = idx & (E_ - 1);
    int b = (idx >> 8) & (B_ - 1);
    int s = idx >> 12;
    int tok = ids[b * S_ + s];
    x[idx] = (bf16_t)emb[(size_t)tok * E_ + e];
}

// ---------------------------------------------------------------------------
// f32 -> bf16 conversion (layer input for layers 1,2)
// ---------------------------------------------------------------------------
__global__ void k_cvt(const float* __restrict__ src, bf16_t* __restrict__ dst, int n) {
    int idx = blockIdx.x * blockDim.x + threadIdx.x;
    if (idx < n) dst[idx] = (bf16_t)src[idx];
}

// ---------------------------------------------------------------------------
// Pack W (row-major [N][K], used as B = W^T of shape KxN) into per-lane
// WMMA bf16 B fragments. Fragment (kt,nt): lane l holds N = nt*16+(l&15),
// halfword h holds K = kt*32 + (l>=16 ? 8:0) + (h&7) + (h>=8 ? 16:0). Contiguous 32B per lane.
// out[((kt*NT+nt)*32 + lane)*16 + h]
// ---------------------------------------------------------------------------
__global__ void k_pack_b(const float* __restrict__ W, bf16_t* __restrict__ out,
                         int K, int N) {
    int idx = blockIdx.x * blockDim.x + threadIdx.x;
    if (idx >= K * N) return;
    int h    = idx & 15;
    int lane = (idx >> 4) & 31;
    int frag = idx >> 9;
    int NT = N >> 4;
    int nt = frag % NT;
    int kt = frag / NT;
    int k = kt * 32 + ((lane >= 16) ? 8 : 0) + (h & 7) + ((h >= 8) ? 16 : 0);
    int n = nt * 16 + (lane & 15);
    out[idx] = (bf16_t)W[(size_t)n * K + k];
}

// ---------------------------------------------------------------------------
// bf16 WMMA GEMM, 64x64 output tile per wave (4x4 WMMA register blocking):
// C[M,N] = A[M,K](row-major bf16) @ Bp(packed) + bias1 + bias2
// Per K-chunk (32): 4 A-frag loads + 4 B-frag loads -> 16 WMMAs.
// ---------------------------------------------------------------------------
__global__ __launch_bounds__(256) void k_gemm64(const bf16_t* __restrict__ A,
                                                const bf16_t* __restrict__ Bp,
                                                const float* __restrict__ bias1,
                                                const float* __restrict__ bias2,
                                                float* __restrict__ C,
                                                int M, int N, int K) {
    int gtid = blockIdx.x * blockDim.x + threadIdx.x;
    int wave = gtid >> 5;
    int lane = gtid & 31;
    int NT = N >> 4, KT = K >> 5;
    int NT64 = N >> 6, MT64 = M >> 6;
    if (wave >= MT64 * NT64) return;        // wave-uniform
    int mt64 = wave / NT64;
    int nt64 = wave % NT64;

    v8f acc[4][4];
#pragma unroll
    for (int mi = 0; mi < 4; ++mi)
#pragma unroll
        for (int ni = 0; ni < 4; ++ni)
            acc[mi][ni] = (v8f){0.f, 0.f, 0.f, 0.f, 0.f, 0.f, 0.f, 0.f};

    const int rowBase = mt64 * 64 + (lane & 15);
    const int kA = (lane >= 16) ? 8 : 0;    // A-frag K sub-offset per ISA layout

    for (int kt = 0; kt < KT; ++kt) {
        v16bf a[4];
#pragma unroll
        for (int mi = 0; mi < 4; ++mi) {
            const bf16_t* ar = A + (size_t)(rowBase + mi * 16) * K + kt * 32 + kA;
            v8bf lo = *(const v8bf*)ar;
            v8bf hi = *(const v8bf*)(ar + 16);
#pragma unroll
            for (int i = 0; i < 8; ++i) { a[mi][i] = lo[i]; a[mi][i + 8] = hi[i]; }
        }
        v16bf b[4];
#pragma unroll
        for (int ni = 0; ni < 4; ++ni)
            b[ni] = *(const v16bf*)(Bp +
                     ((size_t)(kt * NT + nt64 * 4 + ni) * 32 + lane) * 16);
#pragma unroll
        for (int mi = 0; mi < 4; ++mi)
#pragma unroll
            for (int ni = 0; ni < 4; ++ni)
                acc[mi][ni] = __builtin_amdgcn_wmma_f32_16x16x32_bf16(
                    false, a[mi], false, b[ni], (short)0, acc[mi][ni], false, false);
    }

    float bv[4];
#pragma unroll
    for (int ni = 0; ni < 4; ++ni) {
        int n = nt64 * 64 + ni * 16 + (lane & 15);
        float v = 0.f;
        if (bias1) v += bias1[n];
        if (bias2) v += bias2[n];
        bv[ni] = v;
    }
#pragma unroll
    for (int mi = 0; mi < 4; ++mi) {
        int m0 = mt64 * 64 + mi * 16 + ((lane >= 16) ? 8 : 0);
#pragma unroll
        for (int ni = 0; ni < 4; ++ni) {
            int n = nt64 * 64 + ni * 16 + (lane & 15);
#pragma unroll
            for (int r = 0; r < 8; ++r)
                C[(size_t)(m0 + r) * N + n] = acc[mi][ni][r] + bv[ni];
        }
    }
}

// ---------------------------------------------------------------------------
// Persistent LSTM recurrence, one block per direction (grid=2, 512 threads).
// gates[t] = gx[t] + h @ Whh^T ; h,c updated in-place. h kept in LDS in
// pre-permuted WMMA A-fragment layout (bf16). Whh streamed from L2 each step
// (packed B fragments). Wave w owns N-tiles {w, w+16, w+32, w+48} so each
// lane holds i/f/g/o for the same (m, j=w*16+(lane&15)).
// out: f32 [S][B][2H], forward -> cols [0,H), backward -> cols [H,2H).
// ---------------------------------------------------------------------------
__global__ __launch_bounds__(512) void k_lstm(const float* __restrict__ gxF,
                                              const float* __restrict__ gxB,
                                              const bf16_t* __restrict__ WhhFp,
                                              const bf16_t* __restrict__ WhhBp,
                                              float* __restrict__ out) {
    const int dir = blockIdx.x;
    const float*  gx   = dir ? gxB   : gxF;
    const bf16_t* Whhp = dir ? WhhBp : WhhFp;
    const int colOff = dir * H_;

    __shared__ __align__(32) unsigned short hbuf[8 * 32 * 16]; // 8 KB: A frags of h

    const int tid  = threadIdx.x;
    const int wave = tid >> 5;
    const int lane = tid & 31;
    const int laneN  = lane & 15;
    const int laneHi = lane >> 4;
    const int j = wave * 16 + laneN;       // hidden unit owned by this lane

    for (int i = tid; i < 8 * 32 * 16; i += 512) hbuf[i] = 0;
    __syncthreads();

    float c[8];
#pragma unroll
    for (int r = 0; r < 8; ++r) c[r] = 0.f;

    const v8f vzero = {0.f, 0.f, 0.f, 0.f, 0.f, 0.f, 0.f, 0.f};

    for (int step = 0; step < S_; ++step) {
        const int t = dir ? (S_ - 1 - step) : step;

        // Pull next step's gx slice (64 KB) toward the WGP while WMMAs run.
        if (step + 1 < S_) {
            const int tn = dir ? (t - 1) : (t + 1);
            const float* pf = gx + (size_t)tn * 16 * N4H + (size_t)tid * 32;
            __builtin_prefetch(pf, 0, 1);
        }

        v8f acc0 = vzero, acc1 = vzero, acc2 = vzero, acc3 = vzero;
#pragma unroll
        for (int kt = 0; kt < 8; ++kt) {
            v16bf a = *(const v16bf*)(&hbuf[(kt * 32 + lane) * 16]);
            const size_t fb = (size_t)kt * 64;
            v16bf b0 = *(const v16bf*)(Whhp + ((fb + wave +  0) * 32 + lane) * 16);
            v16bf b1 = *(const v16bf*)(Whhp + ((fb + wave + 16) * 32 + lane) * 16);
            v16bf b2 = *(const v16bf*)(Whhp + ((fb + wave + 32) * 32 + lane) * 16);
            v16bf b3 = *(const v16bf*)(Whhp + ((fb + wave + 48) * 32 + lane) * 16);
            acc0 = __builtin_amdgcn_wmma_f32_16x16x32_bf16(false, a, false, b0, (short)0, acc0, false, false);
            acc1 = __builtin_amdgcn_wmma_f32_16x16x32_bf16(false, a, false, b1, (short)0, acc1, false, false);
            acc2 = __builtin_amdgcn_wmma_f32_16x16x32_bf16(false, a, false, b2, (short)0, acc2, false, false);
            acc3 = __builtin_amdgcn_wmma_f32_16x16x32_bf16(false, a, false, b3, (short)0, acc3, false, false);
        }
        __syncthreads();   // all waves done reading hbuf of step t

        const float* gxt = gx + (size_t)t * 16 * N4H;
#pragma unroll
        for (int r = 0; r < 8; ++r) {
            const int m = r + 8 * laneHi;
            const float* grow = gxt + (size_t)m * N4H;
            float gi = acc0[r] + grow[j];
            float gf = acc1[r] + grow[256 + j];
            float gg = acc2[r] + grow[512 + j];
            float go = acc3[r] + grow[768 + j];
            float is = 1.f / (1.f + expf(-gi));
            float fs = 1.f / (1.f + expf(-gf));
            float os = 1.f / (1.f + expf(-go));
            c[r] = fs * c[r] + is * tanhf(gg);
            float h = os * tanhf(c[r]);
            out[(size_t)(t * 16 + m) * (2 * H_) + colOff + j] = h;
            // write h into hbuf in WMMA A-fragment layout (bf16)
            int kt = j >> 5, kk = j & 31;
            int l2 = m + (((kk & 15) >= 8) ? 16 : 0);
            int hh = ((kk >= 16) ? 8 : 0) + (kk & 7);
            hbuf[(kt * 32 + l2) * 16 + hh] =
                __builtin_bit_cast(unsigned short, (bf16_t)h);
        }
        __syncthreads();   // hbuf of step t+1 ready
    }
}

// ---------------------------------------------------------------------------
// Edge feature gather: feat[(b*S+s)][0:512]=lstm[b,src], [512:1024]=lstm[b,tgt]
// ---------------------------------------------------------------------------
__global__ void k_feat(const float* __restrict__ hs,
                       const int* __restrict__ edges,
                       bf16_t* __restrict__ feat) {
    int idx = blockIdx.x * blockDim.x + threadIdx.x;
    if (idx >= B_ * S_ * 1024) return;
    int cdim = idx & 1023;
    int s = (idx >> 10) & (S_ - 1);
    int b = idx >> 20;
    int which = cdim >> 9;
    int cc = cdim & 511;
    int e = edges[((size_t)(b * S_ + s)) * 2 + which];
    feat[idx] = (bf16_t)hs[((size_t)e * 16 + b) * 512 + cc];
}

// ---------------------------------------------------------------------------
// score = relu(hid) @ W2 + b2;  out[b,s,:] = (lstm_out+score) @ cls_W^T + cls_b
// ---------------------------------------------------------------------------
__global__ void k_final(const float* __restrict__ hs,
                        const float* __restrict__ hid,
                        const float* __restrict__ W2, const float* __restrict__ b2,
                        const float* __restrict__ clsW, const float* __restrict__ clsb,
                        float* __restrict__ out) {
    int idx = blockIdx.x * blockDim.x + threadIdx.x;   // over B*S, row = b*S+s
    if (idx >= B_ * S_) return;
    int b = idx >> 10;
    int s = idx & (S_ - 1);
    const float* hrow = hid + (size_t)idx * 256;
    float sc = b2[0];
    for (int k = 0; k < 256; ++k) sc += fmaxf(hrow[k], 0.f) * W2[k];
    const float* lrow = hs + (size_t)(s * 16 + b) * 512;
    for (int t = 0; t < 3; ++t) {
        float acc = clsb[t];
        for (int k = 0; k < 512; ++k) acc += (lrow[k] + sc) * clsW[t * 512 + k];
        out[(size_t)idx * 3 + t] = acc;
    }
}

// ---------------------------------------------------------------------------
extern "C" void kernel_launch(void* const* d_in, const int* in_sizes, int n_in,
                              void* d_out, int out_size, void* d_ws, size_t ws_size,
                              hipStream_t stream) {
    const int*   ids   = (const int*)d_in[0];
    const int*   edges = (const int*)d_in[1];
    const float* emb   = (const float*)d_in[2];
    // lstm_params flattened: layer-major, dir-major, (W_ih, W_hh, b_ih, b_hh)
    auto LW = [&](int l, int d, int which) -> const float* {
        return (const float*)d_in[3 + l * 8 + d * 4 + which];
    };
    const float* eW1  = (const float*)d_in[27];
    const float* eb1  = (const float*)d_in[28];
    const float* eW2  = (const float*)d_in[29];
    const float* eb2  = (const float*)d_in[30];
    const float* clsW = (const float*)d_in[31];
    const float* clsb = (const float*)d_in[32];

    char* ws = (char*)d_ws;
    size_t off = 0;
    auto alloc = [&](size_t bytes) -> void* {
        void* p = ws + off;
        off += (bytes + 255) & ~(size_t)255;
        return p;
    };
    bf16_t* xin   = (bf16_t*)alloc((size_t)MROWS * 512 * 2);   // layer input, bf16
    float*  hsA   = (float*)alloc((size_t)MROWS * 512 * 4);
    float*  hsB   = (float*)alloc((size_t)MROWS * 512 * 4);
    float*  gxF   = (float*)alloc((size_t)MROWS * N4H * 4);    // 64 MB, reused as feat
    float*  gxB   = (float*)alloc((size_t)MROWS * N4H * 4);    // 64 MB, reused as hid
    bf16_t* pWihF = (bf16_t*)alloc((size_t)512 * N4H * 2);
    bf16_t* pWihB = (bf16_t*)alloc((size_t)512 * N4H * 2);
    bf16_t* pWhhF = (bf16_t*)alloc((size_t)H_ * N4H * 2);
    bf16_t* pWhhB = (bf16_t*)alloc((size_t)H_ * N4H * 2);
    bf16_t* pW1   = (bf16_t*)alloc((size_t)N4H * H_ * 2);

    // layer 0 input: embedding gather -> bf16 [S][B][E]
    k_embed<<<(S_ * B_ * E_ + 255) / 256, 256, 0, stream>>>(ids, emb, xin);

    const float* hsPrev = nullptr;
    for (int l = 0; l < 3; ++l) {
        const int K = (l == 0) ? E_ : 512;
        if (l > 0) {
            int n = MROWS * 512;
            k_cvt<<<(n + 255) / 256, 256, 0, stream>>>(hsPrev, xin, n);
        }
        // pack weights (B fragments)
        k_pack_b<<<(K * N4H + 255) / 256, 256, 0, stream>>>(LW(l, 0, 0), pWihF, K, N4H);
        k_pack_b<<<(K * N4H + 255) / 256, 256, 0, stream>>>(LW(l, 1, 0), pWihB, K, N4H);
        k_pack_b<<<(H_ * N4H + 255) / 256, 256, 0, stream>>>(LW(l, 0, 1), pWhhF, H_, N4H);
        k_pack_b<<<(H_ * N4H + 255) / 256, 256, 0, stream>>>(LW(l, 1, 1), pWhhB, H_, N4H);
        // gx = x @ Wih^T + bih + bhh   (M=16384, N=1024), 64x64 tile per wave
        int waves = (MROWS / 64) * (N4H / 64);       // 4096
        int blocks = waves * 32 / 256;               // 512
        k_gemm64<<<blocks, 256, 0, stream>>>(xin, pWihF, LW(l, 0, 2), LW(l, 0, 3),
                                             gxF, MROWS, N4H, K);
        k_gemm64<<<blocks, 256, 0, stream>>>(xin, pWihB, LW(l, 1, 2), LW(l, 1, 3),
                                             gxB, MROWS, N4H, K);
        float* hsOut = (l & 1) ? hsB : hsA;
        k_lstm<<<2, 512, 0, stream>>>(gxF, gxB, pWhhF, pWhhB, hsOut);
        hsPrev = hsOut;
    }

    // edge MLP: feat gather -> GEMM(hid) -> score + classifier
    bf16_t* feat = (bf16_t*)gxF;   // 32 MB, reuse gx region
    float*  hid  = (float*)gxB;    // 16 MB, reuse gx region
    k_pack_b<<<(N4H * H_ + 255) / 256, 256, 0, stream>>>(eW1, pW1, N4H, H_);
    int nf = B_ * S_ * 1024;
    k_feat<<<(nf + 255) / 256, 256, 0, stream>>>(hsPrev, edges, feat);
    int waves2 = (MROWS / 64) * (H_ / 64);           // 1024
    k_gemm64<<<waves2 * 32 / 256, 256, 0, stream>>>(feat, pW1, eb1, nullptr,
                                                    hid, MROWS, H_, N4H);
    k_final<<<(B_ * S_ + 255) / 256, 256, 0, stream>>>(hsPrev, hid, eW2, eb2,
                                                       clsW, clsb, (float*)d_out);
}
